// GNNClassifier_67499706024329
// MI455X (gfx1250) — compile-verified, matrix operand order checked by reading, not run on hardware
//
#include <hip/hip_runtime.h>
#include <hip/hip_bf16.h>
#include <math.h>

typedef __attribute__((ext_vector_type(16))) _Float16 v16h;
typedef __attribute__((ext_vector_type(8)))  float    v8f;
typedef __attribute__((ext_vector_type(4)))  _Float16 v4h;

#define HD 64
#define NWAVES 8   // waves per 256-thread block; each wave owns a 16-row strip

// K index of element e (0..15) of a 16-bit 16x32 WMMA fragment for this lane.
__device__ __forceinline__ int frag_k(int lane, int e) {
  int p = e >> 1, q = e & 1;
  return ((p & 4) << 2) + ((lane & 16) >> 1) + ((p & 3) << 1) + q;
}

// A-fragment: two b128 loads from a row-major f16 row (halves [base..base+7], [base+16..base+23])
__device__ __forceinline__ v16h load_a(const _Float16* __restrict__ row, int lane, int c) {
  int base = c * 32 + ((lane & 16) >> 1);
  v16h r;
  *(int4*)&r       = *(const int4*)(row + base);
  *((int4*)&r + 1) = *(const int4*)(row + base + 16);
  return r;
}

// B-fragment: 16 contiguous halves per lane from a pre-swizzled fragment buffer
__device__ __forceinline__ v16h load_b(const _Float16* __restrict__ frag, int tile, int c, int lane) {
  const _Float16* p = frag + ((((tile << 1) + c) << 5) + lane) * 16;
  v16h r;
  *(int4*)&r       = *(const int4*)p;
  *((int4*)&r + 1) = *(const int4*)(p + 8);
  return r;
}

// ---------------- weight pre-swizzle: B[k][col] = W[k][col] (row-major KxC) ----------------
__global__ void prep_b_direct(const float* __restrict__ W, _Float16* __restrict__ out, int ncols) {
  int i = blockIdx.x * blockDim.x + threadIdx.x;
  int total = (ncols >> 4) * 1024;          // tiles * 2 chunks * 32 lanes * 16 halves
  if (i >= total) return;
  int e    = i & 15;
  int lane = (i >> 4) & 31;
  int c    = (i >> 9) & 1;
  int t    = i >> 10;
  int col = t * 16 + (lane & 15);
  int k   = c * 32 + frag_k(lane, e);
  out[i] = (_Float16)W[k * ncols + col];
}

// ---------------- weight pre-swizzle: B[k][col] = W[col][k] (W row-major Cx64, transposed use) ----
__global__ void prep_b_trans(const float* __restrict__ W, _Float16* __restrict__ out, int ncols) {
  int i = blockIdx.x * blockDim.x + threadIdx.x;
  int total = (ncols >> 4) * 1024;
  if (i >= total) return;
  int e    = i & 15;
  int lane = (i >> 4) & 31;
  int c    = (i >> 9) & 1;
  int t    = i >> 10;
  int col = t * 16 + (lane & 15);
  int k   = c * 32 + frag_k(lane, e);
  out[i] = (_Float16)W[col * HD + k];
}

// ---------------- embedding gather: fills x (f32) and xh (f16), zero pad rows ----------------
__global__ void embed_kernel(const int* __restrict__ tokens, const float* __restrict__ emb,
                             float* __restrict__ x, _Float16* __restrict__ xh, int N, int Npad) {
  int idx = blockIdx.x * blockDim.x + threadIdx.x;
  if (idx >= Npad * 16) return;
  int node = idx >> 4, part = idx & 15;
  float4 v = make_float4(0.f, 0.f, 0.f, 0.f);
  if (node < N) v = ((const float4*)emb)[(size_t)tokens[node] * 16 + part];
  ((float4*)x)[idx] = v;
  v4h h;
  h[0] = (_Float16)v.x; h[1] = (_Float16)v.y; h[2] = (_Float16)v.z; h[3] = (_Float16)v.w;
  ((v4h*)xh)[idx] = h;
}

__global__ void zero_kernel(float* __restrict__ p, int n) {
  int i = blockIdx.x * blockDim.x + threadIdx.x;
  if (i < n) p[i] = 0.0f;
}

// ---------------- f32 -> f16 convert (agg -> aggh) ----------------
__global__ void cvt_kernel(const float* __restrict__ in, _Float16* __restrict__ out, int n4) {
  int i = blockIdx.x * blockDim.x + threadIdx.x;
  if (i >= n4) return;
  float4 v = ((const float4*)in)[i];
  v4h h;
  h[0] = (_Float16)v.x; h[1] = (_Float16)v.y; h[2] = (_Float16)v.z; h[3] = (_Float16)v.w;
  ((v4h*)out)[i] = h;
}

// ---------------- mh = xh @ W (Npad x 64 @ 64 x 64), WMMA f16, f16 output ----------------
__global__ void __launch_bounds__(256) transform_kernel(
    const _Float16* __restrict__ xh, const _Float16* __restrict__ wfrag,
    _Float16* __restrict__ mh) {
  int tid = threadIdx.x, wave = tid >> 5, lane = tid & 31;
  int row0 = (blockIdx.x * NWAVES + wave) * 16;
  int n = lane & 15;
  const _Float16* xr = xh + (size_t)(row0 + n) * HD;
  v16h a0 = load_a(xr, lane, 0);
  v16h a1 = load_a(xr, lane, 1);
  int rbase = row0 + ((lane & 16) >> 1);
#pragma unroll
  for (int t = 0; t < 4; ++t) {
    v16h b0 = load_b(wfrag, t, 0, lane);
    v16h b1 = load_b(wfrag, t, 1, lane);
    v8f acc = {};
    acc = __builtin_amdgcn_wmma_f32_16x16x32_f16(false, a0, false, b0, (short)0, acc, false, false);
    acc = __builtin_amdgcn_wmma_f32_16x16x32_f16(false, a1, false, b1, (short)0, acc, false, false);
    _Float16* mp = mh + (size_t)rbase * HD + t * 16 + n;
#pragma unroll
    for (int v = 0; v < 8; ++v) mp[v * HD] = (_Float16)acc[v];
  }
}

// ---------------- edge scatter: agg[dst] += (f32)mh[src] ----------------
__global__ void scatter_kernel(const _Float16* __restrict__ mh, const int* __restrict__ src,
                               const int* __restrict__ dst, float* __restrict__ agg, int E) {
  int idx = blockIdx.x * blockDim.x + threadIdx.x;
  if (idx >= E * 8) return;
  int e = idx >> 3, part = idx & 7;
  int s = src[e], d = dst[e];
  int4 raw = ((const int4*)(mh + (size_t)s * HD))[part];   // 8 halves
  const _Float16* hp = (const _Float16*)&raw;
  float* ap = agg + (size_t)d * HD + part * 8;
#pragma unroll
  for (int i = 0; i < 8; ++i) unsafeAtomicAdd(ap + i, (float)hp[i]);
}

// ---------------- fused GRU cell: x = GRU(agg, x); updates x (f32) and xh (f16) ----------------
__global__ void __launch_bounds__(256) gru_kernel(
    const _Float16* __restrict__ aggh, const _Float16* __restrict__ xh,
    const _Float16* __restrict__ ihfrag, const _Float16* __restrict__ hhfrag,
    const float* __restrict__ b_ih, const float* __restrict__ b_hh,
    float* __restrict__ x, _Float16* __restrict__ xh_out) {
  int tid = threadIdx.x, wave = tid >> 5, lane = tid & 31;
  int row0 = (blockIdx.x * NWAVES + wave) * 16;
  int n = lane & 15;
  const _Float16* ar = aggh + (size_t)(row0 + n) * HD;
  const _Float16* xr = xh   + (size_t)(row0 + n) * HD;
  v16h aA0 = load_a(ar, lane, 0);
  v16h aA1 = load_a(ar, lane, 1);
  v16h aX0 = load_a(xr, lane, 0);
  v16h aX1 = load_a(xr, lane, 1);

  float rg[4][8], zg[4][8], gin[4][8], ghn[4][8];

#pragma unroll
  for (int g = 0; g < 3; ++g) {            // gates in (r, z, n) order
#pragma unroll
    for (int t = 0; t < 4; ++t) {
      int tt = g * 4 + t;                  // global 16-col tile of the 192-wide gate output
      v16h bi0 = load_b(ihfrag, tt, 0, lane);
      v16h bi1 = load_b(ihfrag, tt, 1, lane);
      v16h bh0 = load_b(hhfrag, tt, 0, lane);
      v16h bh1 = load_b(hhfrag, tt, 1, lane);
      v8f ci = {};
      ci = __builtin_amdgcn_wmma_f32_16x16x32_f16(false, aA0, false, bi0, (short)0, ci, false, false);
      ci = __builtin_amdgcn_wmma_f32_16x16x32_f16(false, aA1, false, bi1, (short)0, ci, false, false);
      v8f ch = {};
      ch = __builtin_amdgcn_wmma_f32_16x16x32_f16(false, aX0, false, bh0, (short)0, ch, false, false);
      ch = __builtin_amdgcn_wmma_f32_16x16x32_f16(false, aX1, false, bh1, (short)0, ch, false, false);
      float bi = b_ih[tt * 16 + n];
      float bh = b_hh[tt * 16 + n];
#pragma unroll
      for (int v = 0; v < 8; ++v) {
        float gi_v = ci[v] + bi;
        float gh_v = ch[v] + bh;
        if (g == 0)      rg[t][v] = 1.0f / (1.0f + __expf(-(gi_v + gh_v)));
        else if (g == 1) zg[t][v] = 1.0f / (1.0f + __expf(-(gi_v + gh_v)));
        else           { gin[t][v] = gi_v; ghn[t][v] = gh_v; }
      }
    }
  }

  int rbase = row0 + ((lane & 16) >> 1);
#pragma unroll
  for (int t = 0; t < 4; ++t) {
    float*    xp  = x      + (size_t)rbase * HD + t * 16 + n;
    _Float16* xhp = xh_out + (size_t)rbase * HD + t * 16 + n;
#pragma unroll
    for (int v = 0; v < 8; ++v) {
      float nv = tanhf(gin[t][v] + rg[t][v] * ghn[t][v]);
      float ho = xp[v * HD];
      float hn = (1.0f - zg[t][v]) * nv + zg[t][v] * ho;
      xp[v * HD]  = hn;
      xhp[v * HD] = (_Float16)hn;
    }
  }
}

// ---------------- relu + mean-pool accumulation ----------------
__global__ void pool_kernel(const float* __restrict__ x, const int* __restrict__ batch,
                            float* __restrict__ sums, float* __restrict__ cnts, int N) {
  int idx = blockIdx.x * blockDim.x + threadIdx.x;
  if (idx >= N * 16) return;
  int node = idx >> 4, part = idx & 15;
  int b = batch[node];
  float4 v = ((const float4*)x)[idx];
  float* sp = sums + (size_t)b * HD + part * 4;
  unsafeAtomicAdd(sp + 0, fmaxf(v.x, 0.0f));
  unsafeAtomicAdd(sp + 1, fmaxf(v.y, 0.0f));
  unsafeAtomicAdd(sp + 2, fmaxf(v.z, 0.0f));
  unsafeAtomicAdd(sp + 3, fmaxf(v.w, 0.0f));
  if (part == 0) unsafeAtomicAdd(&cnts[b], 1.0f);
}

// ---------------- MLP head: 128 graphs, one thread each ----------------
__global__ void head_kernel(const float* __restrict__ sums, const float* __restrict__ cnts,
                            const float* __restrict__ lin1_w, const float* __restrict__ lin1_b,
                            const float* __restrict__ lout_w, const float* __restrict__ lout_b,
                            float* __restrict__ out, int G) {
  int g = blockIdx.x * blockDim.x + threadIdx.x;
  if (g >= G) return;
  float c = fmaxf(cnts[g], 1.0f);
  float p[HD];
#pragma unroll
  for (int k = 0; k < HD; ++k) p[k] = sums[g * HD + k] / c;
  float h[HD];
  for (int j = 0; j < HD; ++j) {
    float acc = lin1_b[j];
#pragma unroll
    for (int k = 0; k < HD; ++k) acc += p[k] * lin1_w[j * HD + k];
    h[j] = fmaxf(acc, 0.0f);
  }
  for (int cl = 0; cl < 2; ++cl) {
    float acc = lout_b[cl];
#pragma unroll
    for (int k = 0; k < HD; ++k) acc += h[k] * lout_w[cl * HD + k];
    out[g * 2 + cl] = acc;
  }
}

extern "C" void kernel_launch(void* const* d_in, const int* in_sizes, int n_in,
                              void* d_out, int out_size, void* d_ws, size_t ws_size,
                              hipStream_t stream) {
  const int*   tokens = (const int*)d_in[0];
  const int*   edges  = (const int*)d_in[1];
  const int*   batch  = (const int*)d_in[2];
  const float* emb    = (const float*)d_in[3];
  const float* ggnn_w = (const float*)d_in[4];
  const float* w_ih   = (const float*)d_in[5];
  const float* w_hh   = (const float*)d_in[6];
  const float* b_ih   = (const float*)d_in[7];
  const float* b_hh   = (const float*)d_in[8];
  const float* lin1_w = (const float*)d_in[9];
  const float* lin1_b = (const float*)d_in[10];
  const float* lout_w = (const float*)d_in[11];
  const float* lout_b = (const float*)d_in[12];
  float* out = (float*)d_out;

  const int N = in_sizes[0];
  const int E = in_sizes[1] / 2;
  const int G = 128;
  const int Npad = (N + 127) & ~127;           // full 128-row blocks, no tail guards
  const int* src = edges;
  const int* dst = edges + E;

  // workspace layout (f32 region then f16 region; all 128B-aligned)
  float* x    = (float*)d_ws;                  // Npad*64
  float* agg  = x    + (size_t)Npad * HD;      // Npad*64
  float* sums = agg  + (size_t)Npad * HD;      // G*64
  float* cnts = sums + (size_t)G * HD;         // G
  _Float16* xh     = (_Float16*)(cnts + G);    // Npad*64
  _Float16* aggh   = xh   + (size_t)Npad * HD; // Npad*64
  _Float16* mh     = aggh + (size_t)Npad * HD; // Npad*64
  _Float16* gfrag  = mh   + (size_t)Npad * HD; // 3*64*64 swizzled transform weights
  _Float16* ihfrag = gfrag  + 3 * HD * HD;     // 192*64
  _Float16* hhfrag = ihfrag + 192 * HD;        // 192*64

  // one-time (per launch) weight swizzle into fragment-major f16
  for (int step = 0; step < 3; ++step)
    prep_b_direct<<<(4 * 1024 + 255) / 256, 256, 0, stream>>>(
        ggnn_w + step * HD * HD, gfrag + step * HD * HD, HD);
  prep_b_trans<<<(12 * 1024 + 255) / 256, 256, 0, stream>>>(w_ih, ihfrag, 192);
  prep_b_trans<<<(12 * 1024 + 255) / 256, 256, 0, stream>>>(w_hh, hhfrag, 192);

  const int nf16p = Npad * 16;
  embed_kernel<<<(nf16p + 255) / 256, 256, 0, stream>>>(tokens, emb, x, xh, N, Npad);

  const int gemm_blocks = Npad / (16 * NWAVES);
  for (int step = 0; step < 3; ++step) {
    transform_kernel<<<gemm_blocks, 256, 0, stream>>>(xh, gfrag + step * HD * HD, mh);
    zero_kernel<<<(Npad * HD + 255) / 256, 256, 0, stream>>>(agg, Npad * HD);
    scatter_kernel<<<(E * 8 + 255) / 256, 256, 0, stream>>>(mh, src, dst, agg, E);
    cvt_kernel<<<(nf16p + 255) / 256, 256, 0, stream>>>(agg, aggh, nf16p);
    gru_kernel<<<gemm_blocks, 256, 0, stream>>>(aggh, xh, ihfrag, hhfrag, b_ih, b_hh, x, xh);
  }

  zero_kernel<<<(G * HD + G + 255) / 256, 256, 0, stream>>>(sums, G * HD + G);
  pool_kernel<<<(N * 16 + 255) / 256, 256, 0, stream>>>(x, batch, sums, cnts, N);
  head_kernel<<<1, 128, 0, stream>>>(sums, cnts, lin1_w, lin1_b, lout_w, lout_b, out, G);
}
